// get_model_30760555774123
// MI455X (gfx1250) — compile-verified, matrix operand order checked by reading, not run on hardware
//
#include <hip/hip_runtime.h>
#include <hip/hip_bf16.h>

#define EPSF   1e-6f
#define SLOPEF 0.2f
#define BNEPSF 1e-5f
#define BB 4           // batch

typedef __bf16 bf16;
typedef __attribute__((ext_vector_type(16))) __bf16 v16bf;
typedef __attribute__((ext_vector_type(8)))  __bf16 v8bf;
typedef __attribute__((ext_vector_type(8)))  float  v8f;

static inline int ceil16i(int x){ return (x+15)&~15; }
static inline int ceil32i(int x){ return (x+31)&~31; }
static inline int nblk(long long n){ return (int)((n+255)/256); }

// ---------------------------------------------------------------- kernels

__global__ void k_split(const float* __restrict__ xyz, float* __restrict__ coords3,
                        float* __restrict__ pts0, int BN){
  int i = blockIdx.x*blockDim.x + threadIdx.x;
  if (i >= BN) return;
  coords3[(size_t)i*3+0] = xyz[(size_t)i*4+0];
  coords3[(size_t)i*3+1] = xyz[(size_t)i*4+1];
  coords3[(size_t)i*3+2] = xyz[(size_t)i*4+2];
  float w = xyz[(size_t)i*4+3];
  pts0[(size_t)i*3+0] = w; pts0[(size_t)i*3+1] = w; pts0[(size_t)i*3+2] = w;
}

// farthest point sampling: one block (256 thr) per batch, sequential npoint steps
__global__ void k_fps(const float* __restrict__ xyz, int N, int npoint, int* __restrict__ idx){
  int b = blockIdx.x;
  const float* X = xyz + (size_t)b*N*3;
  __shared__ float sv[256];
  __shared__ int   si[256];
  float dist[16];
  int t = threadIdx.x;
  int per = (N + 255)/256;
  for (int i=0;i<16;++i) dist[i] = 1e10f;
  int far = 0;
  for (int s=0;s<npoint;++s){
    if (t==0) idx[(size_t)b*npoint + s] = far;
    float cx=X[(size_t)far*3+0], cy=X[(size_t)far*3+1], cz=X[(size_t)far*3+2];
    float best=-1.f; int bi=0;
    for (int i=0;i<per;++i){
      int p = t + i*256;
      if (p < N){
        float dx=X[(size_t)p*3+0]-cx, dy=X[(size_t)p*3+1]-cy, dz=X[(size_t)p*3+2]-cz;
        float d = dx*dx+dy*dy+dz*dz;
        if (d < dist[i]) dist[i] = d;
        if (dist[i] > best){ best = dist[i]; bi = p; }
      }
    }
    sv[t]=best; si[t]=bi;
    __syncthreads();
    for (int off=128; off>0; off>>=1){
      if (t < off){
        if (sv[t+off] > sv[t] || (sv[t+off]==sv[t] && si[t+off] < si[t])){
          sv[t]=sv[t+off]; si[t]=si[t+off];
        }
      }
      __syncthreads();
    }
    far = si[0];
    __syncthreads();
  }
}

__global__ void k_gather_xyz(const float* __restrict__ xyz, const int* __restrict__ idx,
                             float* __restrict__ out, int N, int S, int total){
  int i = blockIdx.x*blockDim.x + threadIdx.x;
  if (i >= total) return;
  int b = i / S;
  int p = idx[i];
  const float* src = xyz + ((size_t)b*N + p)*3;
  out[(size_t)i*3+0]=src[0]; out[(size_t)i*3+1]=src[1]; out[(size_t)i*3+2]=src[2];
}

__global__ void k_ball(const float* __restrict__ xyz, const float* __restrict__ nxyz,
                       int* __restrict__ idx, int N, int S, int ns, float r2, int total){
  int i = blockIdx.x*blockDim.x + threadIdx.x;
  if (i >= total) return;
  int b = i / S;
  const float* X = xyz + (size_t)b*N*3;
  float qx=nxyz[(size_t)i*3+0], qy=nxyz[(size_t)i*3+1], qz=nxyz[(size_t)i*3+2];
  int* o = idx + (size_t)i*ns;
  int cnt=0, first=0; bool have=false;
  for (int j=0; j<N && cnt<ns; ++j){
    float dx=X[(size_t)j*3+0]-qx, dy=X[(size_t)j*3+1]-qy, dz=X[(size_t)j*3+2]-qz;
    if (dx*dx+dy*dy+dz*dz <= r2){ if(!have){first=j;have=true;} o[cnt++]=j; }
  }
  for (; cnt<ns; ++cnt) o[cnt]=first;
}

// pack weight W (Cout x Cin f32) directly into WMMA B-fragment order:
// Bp[((tnIdx*nkb + kbIdx)*32 + lane)*16 + j] = W[n,k] with
//   n  = tnIdx*16 + (lane&15)
//   kk = ((j>>3)<<4) + (j&7) + ((lane>>4)<<3),  k = kbIdx*32 + kk
__global__ void k_pack_w_frag(const float* __restrict__ W, int Cout, int Cin,
                              bf16* __restrict__ Bp, int Kpad, int Npad){
  long long i = (long long)blockIdx.x*blockDim.x + threadIdx.x;
  if (i >= (long long)Kpad*Npad) return;
  int j    = (int)(i & 15);
  int lane = (int)((i >> 4) & 31);
  long long blk = i >> 9;
  int nkb   = Kpad >> 5;
  int kbIdx = (int)(blk % nkb);
  int tnIdx = (int)(blk / nkb);
  int n  = tnIdx*16 + (lane & 15);
  int kk = ((j >> 3) << 4) + (j & 7) + ((lane >> 4) << 3);
  int k  = kbIdx*32 + kk;
  float v = (k < Cin && n < Cout) ? W[(size_t)n*Cin + k] : 0.f;
  Bp[i] = (bf16)v;
}

// pack dense activations feat (M x C f32) -> A (Mpad x Kpad bf16), zero padded
__global__ void k_pack_act(const float* __restrict__ feat, int M, int C,
                           bf16* __restrict__ A, int Mpad, int Kpad){
  long long i = (long long)blockIdx.x*blockDim.x + threadIdx.x;
  if (i >= (long long)Mpad*Kpad) return;
  int c = (int)(i % Kpad); long long m = i / Kpad;
  float v = (m < M && c < C) ? feat[(size_t)m*C + c] : 0.f;
  A[i] = (bf16)v;
}

// build first-layer A for set-abstraction: rows m = ((b*S+s)*ns+k)*3+i3
// col 0: grouped_xyz - new_xyz; cols 1..Cpts: gathered point channels
__global__ void k_group(const float* __restrict__ xyz, const float* __restrict__ pts,
                        const float* __restrict__ nxyz, const int* __restrict__ idx,
                        bf16* __restrict__ A, int N, int S, int ns, int Cpts,
                        int Kpad, int Mpad){
  long long i = (long long)blockIdx.x*blockDim.x + threadIdx.x;
  if (i >= (long long)Mpad*Kpad) return;
  int c = (int)(i % Kpad); long long m = i / Kpad;
  int i3 = (int)(m % 3); long long g = m / 3;
  int k = (int)(g % ns); long long bs = g / ns;
  float v = 0.f;
  if (bs < (long long)BB*S){
    int b = (int)(bs / S);
    int p = idx[(size_t)bs*ns + k];
    if (c == 0)        v = xyz[((size_t)b*N+p)*3 + i3] - nxyz[(size_t)bs*3 + i3];
    else if (c <= Cpts) v = pts[(((size_t)b*N+p)*Cpts + (c-1))*3 + i3];
  }
  A[i] = (bf16)v;
}

static __device__ inline v16bf loadA(const bf16* p, int kb, int khalf8){
  v8bf x = *(const v8bf*)(p + (size_t)kb*32 + khalf8);
  v8bf y = *(const v8bf*)(p + (size_t)kb*32 + 16 + khalf8);
  return __builtin_shufflevector(x, y, 0,1,2,3,4,5,6,7,8,9,10,11,12,13,14,15);
}
static __device__ inline v16bf loadB(const bf16* p, int kb){
  return *(const v16bf*)(p + (size_t)kb*512);
}
static __device__ inline v8f wmma_bf16(v16bf a, v16bf b, v8f c){
  return __builtin_amdgcn_wmma_f32_16x16x32_bf16(false, a, false, b, (short)0, c, false, false);
}

// C[Mpad x Npad] (f32) = A[Mpad x Kpad] (bf16) * Bp (fragment-packed bf16)
// one wave32 per 32x32 macro-tile, double-buffered over K; Mpad % 32 == 0
__global__ void k_gemm_bf16(const bf16* __restrict__ A, const bf16* __restrict__ Bp,
                            float* __restrict__ C, int Mpad, int Kpad, int Npad,
                            int tiles4, int tilesN2){
  int wave = (blockIdx.x*blockDim.x + threadIdx.x) >> 5;
  if (wave >= tiles4) return;
  int lane = threadIdx.x & 31;
  int tm0 = (wave / tilesN2) << 5;
  int tn0 = (wave % tilesN2) << 5;
  bool has2 = (tn0 + 16) < Npad;
  int r0 = tm0 + (lane & 15);
  int khalf8 = (lane >> 4) << 3;
  int nkb = Kpad >> 5;
  const bf16* Ar0 = A + (size_t)r0 * Kpad;
  const bf16* Ar1 = Ar0 + (size_t)16 * Kpad;
  const bf16* B0  = Bp + (size_t)(tn0 >> 4) * nkb * 512 + (size_t)lane * 16;
  const bf16* B1  = B0 + (size_t)nkb * 512;
  v8f acc00 = {0.f,0.f,0.f,0.f,0.f,0.f,0.f,0.f};
  v8f acc01 = acc00, acc10 = acc00, acc11 = acc00;

  if (has2){
    v16bf a0 = loadA(Ar0,0,khalf8), a1 = loadA(Ar1,0,khalf8);
    v16bf b0 = loadB(B0,0),          b1 = loadB(B1,0);
    for (int kb = 0; kb + 1 < nkb; ++kb){
      v16bf na0 = loadA(Ar0,kb+1,khalf8), na1 = loadA(Ar1,kb+1,khalf8);
      v16bf nb0 = loadB(B0,kb+1),          nb1 = loadB(B1,kb+1);
      acc00 = wmma_bf16(a0,b0,acc00);  acc01 = wmma_bf16(a0,b1,acc01);
      acc10 = wmma_bf16(a1,b0,acc10);  acc11 = wmma_bf16(a1,b1,acc11);
      a0 = na0; a1 = na1; b0 = nb0; b1 = nb1;
    }
    acc00 = wmma_bf16(a0,b0,acc00);  acc01 = wmma_bf16(a0,b1,acc01);
    acc10 = wmma_bf16(a1,b0,acc10);  acc11 = wmma_bf16(a1,b1,acc11);
  } else {
    v16bf a0 = loadA(Ar0,0,khalf8), a1 = loadA(Ar1,0,khalf8);
    v16bf b0 = loadB(B0,0);
    for (int kb = 0; kb + 1 < nkb; ++kb){
      v16bf na0 = loadA(Ar0,kb+1,khalf8), na1 = loadA(Ar1,kb+1,khalf8);
      v16bf nb0 = loadB(B0,kb+1);
      acc00 = wmma_bf16(a0,b0,acc00);
      acc10 = wmma_bf16(a1,b0,acc10);
      a0 = na0; a1 = na1; b0 = nb0;
    }
    acc00 = wmma_bf16(a0,b0,acc00);
    acc10 = wmma_bf16(a1,b0,acc10);
  }

  int rowBase = tm0 + ((lane >> 4) << 3);
  int col0 = tn0 + (lane & 15);
#pragma unroll
  for (int v = 0; v < 8; ++v) C[(size_t)(rowBase+v)*Npad + col0]      = acc00[v];
#pragma unroll
  for (int v = 0; v < 8; ++v) C[(size_t)(rowBase+16+v)*Npad + col0]   = acc10[v];
  if (has2){
    int col1 = col0 + 16;
#pragma unroll
    for (int v = 0; v < 8; ++v) C[(size_t)(rowBase+v)*Npad + col1]    = acc01[v];
#pragma unroll
    for (int v = 0; v < 8; ++v) C[(size_t)(rowBase+16+v)*Npad + col1] = acc11[v];
  }
}

// n[g,o] = ||p[g,:,o]||_2 + EPS   (p rows m=g*3+i3, stride Npad)
__global__ void k_vnnorm(const float* __restrict__ p, float* __restrict__ n,
                         int G, int Npad, int Cout){
  long long i = (long long)blockIdx.x*blockDim.x + threadIdx.x;
  if (i >= (long long)G*Cout) return;
  int o = (int)(i % Cout); long long g = i / Cout;
  const float* pp = p + (size_t)g*3*Npad + o;
  float a=pp[0], b=pp[Npad], c=pp[2*(size_t)Npad];
  n[(size_t)g*Npad + o] = sqrtf(a*a+b*b+c*c) + EPSF;
}

// per-channel mean/var of n over all G groups
__global__ void k_chanstats(const float* __restrict__ n, float* __restrict__ mean,
                            float* __restrict__ var, int G, int Npad){
  int o = blockIdx.x;
  __shared__ float s1[256], s2[256];
  float a=0.f, b=0.f;
  for (int g=threadIdx.x; g<G; g+=256){
    float x = n[(size_t)g*Npad + o];
    a += x; b += x*x;
  }
  s1[threadIdx.x]=a; s2[threadIdx.x]=b;
  __syncthreads();
  for (int off=128; off>0; off>>=1){
    if (threadIdx.x < off){ s1[threadIdx.x]+=s1[threadIdx.x+off]; s2[threadIdx.x]+=s2[threadIdx.x+off]; }
    __syncthreads();
  }
  if (threadIdx.x==0){
    float m = s1[0]/G;
    mean[o] = m;
    var[o]  = s2[0]/G - m*m;
  }
}

// VN nonlinearity: BN on norms, rescale p, leaky projection against d
__global__ void k_vnapply(const float* __restrict__ p, const float* __restrict__ d,
                          const float* __restrict__ n, const float* __restrict__ mean,
                          const float* __restrict__ var, const float* __restrict__ gg,
                          const float* __restrict__ bb, float* __restrict__ out,
                          int G, int Npad, int Cout){
  long long i = (long long)blockIdx.x*blockDim.x + threadIdx.x;
  if (i >= (long long)G*Cout) return;
  int o = (int)(i % Cout); long long g = i / Cout;
  const float* pp = p + (size_t)g*3*Npad + o;
  const float* dd = d + (size_t)g*3*Npad + o;
  float nn = n[(size_t)g*Npad + o];
  float nb = gg[o]*(nn - mean[o])*rsqrtf(var[o] + BNEPSF) + bb[o];
  float f  = nb / nn;
  float p0=pp[0]*f, p1=pp[Npad]*f, p2=pp[2*(size_t)Npad]*f;
  float d0=dd[0], d1=dd[Npad], d2=dd[2*(size_t)Npad];
  float dot = p0*d0+p1*d1+p2*d2;
  float den = d0*d0+d1*d1+d2*d2 + EPSF;
  float t_  = dot/den;
  float q0,q1,q2;
  if (dot >= 0.f){ q0=p0; q1=p1; q2=p2; }
  else { q0=p0-t_*d0; q1=p1-t_*d1; q2=p2-t_*d2; }
  float* oo = out + (size_t)g*3*Cout + o;
  oo[0]              = SLOPEF*p0 + (1.f-SLOPEF)*q0;
  oo[Cout]           = SLOPEF*p1 + (1.f-SLOPEF)*q1;
  oo[2*(size_t)Cout] = SLOPEF*p2 + (1.f-SLOPEF)*q2;
}

// VN max pool (argmax over k of <x, Wpool x>) + mean pool; out (BS, 2C, 3)
__global__ void k_pool(const float* __restrict__ feat, const float* __restrict__ dpool,
                       float* __restrict__ out, int BS, int ns, int C, int Npad){
  long long i = (long long)blockIdx.x*blockDim.x + threadIdx.x;
  if (i >= (long long)BS*C) return;
  int c = (int)(i % C); int bs = (int)(i / C);
  size_t base = (size_t)bs*ns*3;
  float best=-1e30f; int bk=0;
  float m0=0.f, m1=0.f, m2=0.f;
  for (int k=0;k<ns;++k){
    size_t m = base + (size_t)k*3;
    float x0=feat[m*C+c], x1=feat[(m+1)*C+c], x2=feat[(m+2)*C+c];
    float d0=dpool[m*Npad+c], d1=dpool[(m+1)*Npad+c], d2=dpool[(m+2)*Npad+c];
    float dot = x0*d0+x1*d1+x2*d2;
    if (dot > best){ best=dot; bk=k; }
    m0+=x0; m1+=x1; m2+=x2;
  }
  size_t mb = base + (size_t)bk*3;
  float* om = out + ((size_t)bs*2*C + c)*3;
  om[0]=feat[mb*C+c]; om[1]=feat[(mb+1)*C+c]; om[2]=feat[(mb+2)*C+c];
  float* oa = out + ((size_t)bs*2*C + C + c)*3;
  float inv = 1.f/(float)ns;
  oa[0]=m0*inv; oa[1]=m1*inv; oa[2]=m2*inv;
}

// kNN over flattened (C*3) features; one block per (b,s)
__global__ void k_knn(const float* __restrict__ pts, int S, int C, int k, int* __restrict__ idx){
  int bs = blockIdx.x; int b = bs / S; int s = bs % S;
  int D = C*3;
  extern __shared__ float sh[];
  float* cen  = sh;        // D
  float* dist = sh + D;    // S
  const float* P = pts + (size_t)b*S*D;
  for (int j=threadIdx.x; j<D; j+=256) cen[j] = P[(size_t)s*D + j];
  __syncthreads();
  for (int s2=threadIdx.x; s2<S; s2+=256){
    const float* q = P + (size_t)s2*D;
    float acc=0.f;
    for (int j=0;j<D;++j){ float df=q[j]-cen[j]; acc += df*df; }
    dist[s2]=acc;
  }
  __syncthreads();
  __shared__ float rv[256]; __shared__ int ri[256];
  for (int kk=0; kk<k; ++kk){
    float bv=1e30f; int bi=S;
    for (int s2=threadIdx.x; s2<S; s2+=256){
      if (dist[s2] < bv){ bv=dist[s2]; bi=s2; }
    }
    rv[threadIdx.x]=bv; ri[threadIdx.x]=bi;
    __syncthreads();
    for (int off=128; off>0; off>>=1){
      if (threadIdx.x < off){
        if (rv[threadIdx.x+off] < rv[threadIdx.x] ||
            (rv[threadIdx.x+off]==rv[threadIdx.x] && ri[threadIdx.x+off]<ri[threadIdx.x])){
          rv[threadIdx.x]=rv[threadIdx.x+off]; ri[threadIdx.x]=ri[threadIdx.x+off];
        }
      }
      __syncthreads();
    }
    if (threadIdx.x==0){ idx[(size_t)bs*k + kk] = ri[0]; dist[ri[0]] = 1e30f; }
    __syncthreads();
  }
}

// build first-layer A for DGCNN edges: cols [0,C): nbr-center, [C,2C): center
__global__ void k_edge(const float* __restrict__ pts, const int* __restrict__ idx,
                       bf16* __restrict__ A, int S, int C, int k, int Kpad, int Mpad){
  long long i = (long long)blockIdx.x*blockDim.x + threadIdx.x;
  if (i >= (long long)Mpad*Kpad) return;
  int c = (int)(i % Kpad); long long m = i / Kpad;
  int i3 = (int)(m % 3); long long g = m / 3;
  int kk = (int)(g % k); long long bs = g / k;
  float v = 0.f;
  if (bs < (long long)BB*S && c < 2*C){
    int b = (int)(bs / S); int s = (int)(bs % S);
    const float* P = pts + (size_t)b*S*C*3;
    if (c < C){
      int s2 = idx[(size_t)bs*k + kk];
      v = P[((size_t)s2*C + c)*3 + i3] - P[((size_t)s*C + c)*3 + i3];
    } else {
      v = P[((size_t)s*C + (c - C))*3 + i3];
    }
  }
  A[i] = (bf16)v;
}

// mean over k edges -> (BS, C, 3)
__global__ void k_meanpool(const float* __restrict__ feat, float* __restrict__ out,
                           int BS, int k, int C){
  long long i = (long long)blockIdx.x*blockDim.x + threadIdx.x;
  if (i >= (long long)BS*C*3) return;
  int i3 = (int)(i % 3); long long rest = i/3;
  int c = (int)(rest % C); int bs = (int)(rest / C);
  float s = 0.f;
  for (int kk=0; kk<k; ++kk){
    size_t m = ((size_t)bs*k + kk)*3 + i3;
    s += feat[m*C + c];
  }
  out[((size_t)bs*C + c)*3 + i3] = s/(float)k;
}

// head BN stats over rows (B*Np) per channel
__global__ void k_headstats(const float* __restrict__ gm, const float* __restrict__ bias,
                            float* __restrict__ mean, float* __restrict__ var,
                            int rows, int Npad, int Cout){
  int o = blockIdx.x*blockDim.x + threadIdx.x;
  if (o >= Cout) return;
  float s=0.f, ss=0.f;
  for (int m=0;m<rows;++m){
    float x = gm[(size_t)m*Npad + o] + bias[o];
    s += x; ss += x*x;
  }
  float mu = s/rows;
  mean[o]=mu; var[o]=ss/rows - mu*mu;
}

__global__ void k_headapply(const float* __restrict__ gm, const float* __restrict__ bias,
                            const float* __restrict__ mean, const float* __restrict__ var,
                            const float* __restrict__ g, const float* __restrict__ beta,
                            float* __restrict__ emb, int rows, int Npad, int Cout){
  int i = blockIdx.x*blockDim.x + threadIdx.x;
  if (i >= rows*Cout) return;
  int o = i % Cout; int m = i / Cout;
  float x = gm[(size_t)m*Npad + o] + bias[o];
  float y = g[o]*(x - mean[o])*rsqrtf(var[o] + BNEPSF) + beta[o];
  emb[i] = y > 0.f ? y : 0.f;
}

__global__ void k_fcoords(const float* __restrict__ emb, const float* __restrict__ W,
                          const float* __restrict__ b, const float* __restrict__ lastxyz,
                          float* __restrict__ out, int BN){
  int i = blockIdx.x*blockDim.x + threadIdx.x;
  if (i >= BN*30) return;                 // (m, a, m3), APP=10
  int m3 = i % 3; int rest = i/3;
  int a = rest % 10; int m = rest / 10;
  const float* e = emb + (size_t)m*480 + a*48 + m3*16;
  float s = b[0];
  for (int dd=0; dd<16; ++dd) s += e[dd]*W[dd];
  out[i] = s + lastxyz[(size_t)m*3 + m3];
}

__global__ void k_fatoms(const float* __restrict__ emb, const float* __restrict__ W,
                         const float* __restrict__ b, float* __restrict__ out, int BN){
  int i = blockIdx.x*blockDim.x + threadIdx.x;
  if (i >= BN*10) return;                 // (m, a)
  int a = i % 10; int m = i / 10;
  const float* e = emb + (size_t)m*160 + a*16;
  float l[10]; float mx = -1e30f;
  for (int t=0;t<10;++t){
    float s = b[t];
    for (int dd=0; dd<16; ++dd) s += e[dd]*W[t*16+dd];
    l[t]=s; if (s>mx) mx=s;
  }
  float se=0.f;
  for (int t=0;t<10;++t) se += expf(l[t]-mx);
  float lse = mx + logf(se);
  float* o = out + (size_t)i*10;
  for (int t=0;t<10;++t) o[t] = l[t]-lse;
}

__global__ void k_copy(const float* __restrict__ src, float* __restrict__ dst, int n){
  int i = blockIdx.x*blockDim.x + threadIdx.x;
  if (i < n) dst[i]=src[i];
}

// ---------------------------------------------------------------- host helpers

struct VNL { const float *Wf, *Wd, *g, *b; int Cout; };

struct Ctx {
  hipStream_t st;
  bf16 *Apack, *Bmat;
  float *pbuf, *dbuf, *nbuf, *feat, *meanb, *varb;
};

static void run_gemm(Ctx& c, const bf16* A, const bf16* Bm, float* C,
                     int Mpad, int Kpad, int Npad){
  int tilesN2 = ((Npad >> 4) + 1) >> 1;        // pairs of 16-wide N tiles
  int tiles4  = (Mpad >> 5) * tilesN2;         // 32-row macro tiles
  int blocks  = (tiles4 + 7)/8;                // 8 waves/block
  k_gemm_bf16<<<blocks, 256, 0, c.st>>>(A, Bm, C, Mpad, Kpad, Npad, tiles4, tilesN2);
}

static void build_layers(void* const* din, int base, const int* outs, int nl, VNL* L){
  for (int i=0;i<nl;++i){
    L[i].Wf = (const float*)din[base+4*i+0];
    L[i].Wd = (const float*)din[base+4*i+1];
    L[i].g  = (const float*)din[base+4*i+2];
    L[i].b  = (const float*)din[base+4*i+3];
    L[i].Cout = outs[i];
  }
}

// Apack must already hold layer-0 input (Mpad x ceil32(Cin0), bf16, zero-padded)
static void run_vn_stack(Ctx& c, const VNL* L, int nl, int Cin0, int M, int G, int Mpad){
  int Cin = Cin0;
  for (int i=0;i<nl;++i){
    int Cout = L[i].Cout;
    int Kpad = ceil32i(Cin), Npad = ceil16i(Cout);
    k_pack_w_frag<<<nblk((long long)Kpad*Npad),256,0,c.st>>>(L[i].Wf, Cout, Cin, c.Bmat, Kpad, Npad);
    run_gemm(c, c.Apack, c.Bmat, c.pbuf, Mpad, Kpad, Npad);
    k_pack_w_frag<<<nblk((long long)Kpad*Npad),256,0,c.st>>>(L[i].Wd, Cout, Cin, c.Bmat, Kpad, Npad);
    run_gemm(c, c.Apack, c.Bmat, c.dbuf, Mpad, Kpad, Npad);
    k_vnnorm<<<nblk((long long)G*Cout),256,0,c.st>>>(c.pbuf, c.nbuf, G, Npad, Cout);
    k_chanstats<<<Cout,256,0,c.st>>>(c.nbuf, c.meanb, c.varb, G, Npad);
    k_vnapply<<<nblk((long long)G*Cout),256,0,c.st>>>(c.pbuf, c.dbuf, c.nbuf, c.meanb, c.varb,
                                                      L[i].g, L[i].b, c.feat, G, Npad, Cout);
    if (i+1 < nl){
      int Kn = ceil32i(Cout);
      k_pack_act<<<nblk((long long)Mpad*Kn),256,0,c.st>>>(c.feat, M, Cout, c.Apack, Mpad, Kn);
    }
    Cin = Cout;
  }
}

static void sa_stage(Ctx& c, const float* xyzIn, const float* ptsIn, int N, int Cpts,
                     int S, float radius, int ns, const VNL* L, int nl, const float* Wpool,
                     int* idxFps, int* idxBall, float* newXyz, float* ptsOut){
  k_fps<<<BB,256,0,c.st>>>(xyzIn, N, S, idxFps);
  k_gather_xyz<<<nblk(BB*S),256,0,c.st>>>(xyzIn, idxFps, newXyz, N, S, BB*S);
  k_ball<<<nblk(BB*S),256,0,c.st>>>(xyzIn, newXyz, idxBall, N, S, ns, radius*radius, BB*S);
  int G = BB*S*ns, M = G*3, Mpad = ceil32i(M);
  int Cin = 1 + Cpts, Kpad = ceil32i(Cin);
  k_group<<<nblk((long long)Mpad*Kpad),256,0,c.st>>>(xyzIn, ptsIn, newXyz, idxBall,
                                                     c.Apack, N, S, ns, Cpts, Kpad, Mpad);
  run_vn_stack(c, L, nl, Cin, M, G, Mpad);
  int C = L[nl-1].Cout;
  int Kp2 = ceil32i(C), Np2 = ceil16i(C);
  k_pack_act<<<nblk((long long)Mpad*Kp2),256,0,c.st>>>(c.feat, M, C, c.Apack, Mpad, Kp2);
  k_pack_w_frag<<<nblk((long long)Kp2*Np2),256,0,c.st>>>(Wpool, C, C, c.Bmat, Kp2, Np2);
  run_gemm(c, c.Apack, c.Bmat, c.dbuf, Mpad, Kp2, Np2);
  k_pool<<<nblk((long long)BB*S*C),256,0,c.st>>>(c.feat, c.dbuf, ptsOut, BB*S, ns, C, Np2);
}

static void dgcnn_stage(Ctx& c, const float* ptsIn, int S, int C, int kk,
                        const VNL* L, int nl, int* idxK, float* ptsOut){
  size_t shmem = (size_t)(C*3 + S)*sizeof(float);
  k_knn<<<BB*S,256,shmem,c.st>>>(ptsIn, S, C, kk, idxK);
  int G = BB*S*kk, M = G*3, Mpad = ceil32i(M);
  int Cin = 2*C, Kpad = ceil32i(Cin);
  k_edge<<<nblk((long long)Mpad*Kpad),256,0,c.st>>>(ptsIn, idxK, c.Apack, S, C, kk, Kpad, Mpad);
  run_vn_stack(c, L, nl, Cin, M, G, Mpad);
  int Cl = L[nl-1].Cout;
  k_meanpool<<<nblk((long long)BB*S*Cl*3),256,0,c.st>>>(c.feat, ptsOut, BB*S, kk, Cl);
}

// ---------------------------------------------------------------- entry

extern "C" void kernel_launch(void* const* d_in, const int* in_sizes, int n_in,
                              void* d_out, int out_size, void* d_ws, size_t ws_size,
                              hipStream_t stream){
  (void)in_sizes; (void)n_in; (void)out_size; (void)ws_size;

  // input leaf indices (setup_inputs insertion order)
  const int IX_XYZ=0, IX_SA1=1, IX_SA1_POOL=13, IX_GNN2=14, IX_SA2=34, IX_SA2_POOL=46;
  const int IX_GNN3=47, IX_SA3=67, IX_SA3_POOL=79, IX_CC=80, IX_CT=84, IX_FC1=88, IX_FC2=90;

  const float* xyz = (const float*)d_in[IX_XYZ];

  // ---- workspace bump allocator
  size_t off = 0;
  auto wsAlloc = [&](size_t bytes)->void*{
    void* p = (char*)d_ws + off;
    off += (bytes + 255) & ~(size_t)255;
    return p;
  };
  bf16* Apack  = (bf16*) wsAlloc(47185920ull*sizeof(bf16));   // 737280 x 64
  bf16* Bmat   = (bf16*) wsAlloc(  983040ull*sizeof(bf16));   // 2048 x 480
  float* pbuf  = (float*)wsAlloc(35389440ull*sizeof(float));  // 737280 x 48
  float* dbuf  = (float*)wsAlloc(35389440ull*sizeof(float));
  float* nbuf  = (float*)wsAlloc(11796480ull*sizeof(float));  // 245760 x 48
  float* feat  = (float*)wsAlloc(30965760ull*sizeof(float));  // 737280 x 42
  float* meanb = (float*)wsAlloc(512*sizeof(float));
  float* varb  = (float*)wsAlloc(512*sizeof(float));
  float* coords3 = (float*)wsAlloc((size_t)BB*4096*3*sizeof(float));
  float* pts0    = (float*)wsAlloc((size_t)BB*4096*3*sizeof(float));
  int*   idxFps  = (int*)  wsAlloc((size_t)BB*1024*sizeof(int));
  int*   idxBall = (int*)  wsAlloc((size_t)BB*1024*60*sizeof(int));
  int*   idxKnn  = (int*)  wsAlloc((size_t)BB*1024*10*sizeof(int));
  float* nx1   = (float*)wsAlloc((size_t)BB*1024*3*sizeof(float));
  float* l1p   = (float*)wsAlloc((size_t)BB*1024*84*3*sizeof(float));
  float* l2f   = (float*)wsAlloc((size_t)BB*1024*42*3*sizeof(float));
  float* nx2   = (float*)wsAlloc((size_t)BB*128*3*sizeof(float));
  float* l2p   = (float*)wsAlloc((size_t)BB*128*170*3*sizeof(float));
  float* l3f   = (float*)wsAlloc((size_t)BB*128*170*3*sizeof(float));
  float* nx3   = (float*)wsAlloc((size_t)BB*10*3*sizeof(float));
  float* lastp = (float*)wsAlloc((size_t)BB*10*682*3*sizeof(float));
  float* emb1  = (float*)wsAlloc((size_t)64*480*sizeof(float));
  float* emb2  = (float*)wsAlloc((size_t)64*160*sizeof(float));

  Ctx c{stream, Apack, Bmat, pbuf, dbuf, nbuf, feat, meanb, varb};

  // ---- split input
  k_split<<<nblk((long long)BB*4096),256,0,stream>>>(xyz, coords3, pts0, BB*4096);

  // ---- SA1: N=4096 -> S=1024, r=0.1, ns=60, layers 2->21->21->42
  {
    const int outs[3] = {21,21,42};
    VNL L[3]; build_layers(d_in, IX_SA1, outs, 3, L);
    sa_stage(c, coords3, pts0, 4096, 1, 1024, 0.1f, 60, L, 3,
             (const float*)d_in[IX_SA1_POOL], idxFps, idxBall, nx1, l1p);
  }
  // ---- GNN2 on l1_points (C=84), k=10, layers 168->85->85->170->170->42
  {
    const int outs[5] = {85,85,170,170,42};
    VNL L[5]; build_layers(d_in, IX_GNN2, outs, 5, L);
    dgcnn_stage(c, l1p, 1024, 84, 10, L, 5, idxKnn, l2f);
  }
  // ---- SA2: N=1024 -> S=128, r=0.3, ns=20, layers 43->42->42->85
  {
    const int outs[3] = {42,42,85};
    VNL L[3]; build_layers(d_in, IX_SA2, outs, 3, L);
    sa_stage(c, nx1, l2f, 1024, 42, 128, 0.3f, 20, L, 3,
             (const float*)d_in[IX_SA2_POOL], idxFps, idxBall, nx2, l2p);
  }
  // ---- GNN3 on l2_points (C=170), k=5, layers 340->170->170->341->341->170
  {
    const int outs[5] = {170,170,341,341,170};
    VNL L[5]; build_layers(d_in, IX_GNN3, outs, 5, L);
    dgcnn_stage(c, l2p, 128, 170, 5, L, 5, idxKnn, l3f);
  }
  // ---- SA3: N=128 -> S=10, r=0.5, ns=8, layers 171->170->170->341
  {
    const int outs[3] = {170,170,341};
    VNL L[3]; build_layers(d_in, IX_SA3, outs, 3, L);
    sa_stage(c, nx2, l3f, 128, 170, 10, 0.5f, 8, L, 3,
             (const float*)d_in[IX_SA3_POOL], idxFps, idxBall, nx3, lastp);
  }

  // ---- heads: lp rows m=(b*Np+n), cols 2046 ; Mpad rounded to 32 -> 64
  const int rows = BB*10, Mpad = 64, Kp = 2048;
  k_pack_act<<<nblk((long long)Mpad*Kp),256,0,stream>>>(lastp, rows, 2046, Apack, Mpad, Kp);

  // conv_coord: 480 channels
  k_pack_w_frag<<<nblk((long long)Kp*480),256,0,stream>>>((const float*)d_in[IX_CC], 480, 2046, Bmat, Kp, 480);
  run_gemm(c, Apack, Bmat, pbuf, Mpad, Kp, 480);
  k_headstats<<<nblk(480),256,0,stream>>>(pbuf, (const float*)d_in[IX_CC+1], meanb, varb, rows, 480, 480);
  k_headapply<<<nblk((long long)rows*480),256,0,stream>>>(pbuf, (const float*)d_in[IX_CC+1], meanb, varb,
      (const float*)d_in[IX_CC+2], (const float*)d_in[IX_CC+3], emb1, rows, 480, 480);

  // conv_type: 160 channels
  k_pack_w_frag<<<nblk((long long)Kp*160),256,0,stream>>>((const float*)d_in[IX_CT], 160, 2046, Bmat, Kp, 160);
  run_gemm(c, Apack, Bmat, dbuf, Mpad, Kp, 160);
  k_headstats<<<nblk(160),256,0,stream>>>(dbuf, (const float*)d_in[IX_CT+1], meanb, varb, rows, 160, 160);
  k_headapply<<<nblk((long long)rows*160),256,0,stream>>>(dbuf, (const float*)d_in[IX_CT+1], meanb, varb,
      (const float*)d_in[IX_CT+2], (const float*)d_in[IX_CT+3], emb2, rows, 160, 160);

  // ---- outputs: [last_xyz 120][coords 1200][atoms 4000]
  float* out = (float*)d_out;
  k_copy<<<1,128,0,stream>>>(nx3, out, 120);
  k_fcoords<<<nblk(rows*30),256,0,stream>>>(emb1, (const float*)d_in[IX_FC1],
      (const float*)d_in[IX_FC1+1], nx3, out + 120, rows);
  k_fatoms<<<nblk(rows*10),256,0,stream>>>(emb2, (const float*)d_in[IX_FC2],
      (const float*)d_in[IX_FC2+1], out + 1320, rows);
}
